// PointerNet_16612933501573
// MI455X (gfx1250) — matrix operand order repaired and checked by reading, not compiled, not run on hardware
//
#include <hip/hip_runtime.h>

// Pointer network for MI455X (gfx1250): bf16 WMMA GEMMs (f32 accumulate),
// 2x4 register-blocked WMMA tiles, bf16 L2-resident ref tensors,
// fused attention/softmax/loss kernels.

static constexpr int Bn  = 256;   // batch
static constexpr int Sn  = 128;   // sequence
static constexpr int En  = 512;   // embed
static constexpr int Hn  = 512;   // hidden
static constexpr int G4H = 4 * Hn;

typedef __attribute__((ext_vector_type(16))) __bf16 v16bf;
typedef __attribute__((ext_vector_type(8)))  __bf16 v8bf;
typedef __attribute__((ext_vector_type(8)))  float  v8f;

__device__ __forceinline__ __bf16 f2bf(float f) {
    unsigned u = __builtin_bit_cast(unsigned, f);
    unsigned r = (u + 0x7fffu + ((u >> 16) & 1u)) >> 16;
    unsigned short s = (unsigned short)r;
    return __builtin_bit_cast(__bf16, s);
}
__device__ __forceinline__ float bf2f(__bf16 b) {
    unsigned short s = __builtin_bit_cast(unsigned short, b);
    unsigned u = ((unsigned)s) << 16;
    return __builtin_bit_cast(float, u);
}
__device__ __forceinline__ float sigmoidf_(float x) { return 1.0f / (1.0f + __expf(-x)); }

__device__ __forceinline__ v16bf load_a_frag(const __bf16* p) {
    v8bf lo = *(const v8bf*)(p);        // K {0-7 | 8-15} of this lane-half
    v8bf hi = *(const v8bf*)(p + 16);   // K {16-23 | 24-31}
    return __builtin_shufflevector(lo, hi, 0,1,2,3,4,5,6,7,8,9,10,11,12,13,14,15);
}
__device__ __forceinline__ v16bf load_b_frag(const __bf16* p) {
    v8bf lo = *(const v8bf*)(p);        // K {0-7 | 8-15} of this column
    v8bf hi = *(const v8bf*)(p + 16);   // K {16-23 | 24-31}
    return __builtin_shufflevector(lo, hi, 0,1,2,3,4,5,6,7,8,9,10,11,12,13,14,15);
}

// ---------------------------------------------------------------------------
// bf16 WMMA GEMM:  C[m,n] = sum_k A[m,k] * W[n,k] (+ bias[n])
// A rows optionally gathered through an index array (embedding lookups fused).
// Block = 128 threads (4 waves). Each wave computes a 32x64 output macro-tile
// (2 M-tiles x 4 N-tiles): A fragments reused 4x, B fragments 2x ->
// 12 b128 loads per 8 v_wmma_f32_16x16x32_bf16.
// ---------------------------------------------------------------------------
__global__ __launch_bounds__(128)
void gemm_bf16_kernel(const __bf16* __restrict__ A, long lda,
                      const int* __restrict__ gather, int gstride,
                      const __bf16* __restrict__ W, long ldw,
                      const float* __restrict__ bias,
                      float* __restrict__ Cf, __bf16* __restrict__ Cb, long ldc,
                      int K)
{
    const int lane = threadIdx.x & 31;
    const int wv   = threadIdx.x >> 5;
    const int m0   = blockIdx.x * 32;
    const int n0   = (blockIdx.y * 4 + wv) * 64;
    const int half = lane >> 4;           // which 16-lane half of the wave
    const int sub  = lane & 15;

    // Two A fragment rows (M tiles m0, m0+16); lane-half picks K sub-chunks
    // {0-7,16-23} / {8-15,24-31} per the 16-bit A layout table.
    int ar0 = m0 + sub, ar1 = m0 + 16 + sub;
    long ai0 = gather ? (long)gather[(long)ar0 * gstride] : (long)ar0;
    long ai1 = gather ? (long)gather[(long)ar1 * gstride] : (long)ar1;
    const __bf16* ap0 = A + ai0 * lda + (half << 3);
    const __bf16* ap1 = A + ai1 * lda + (half << 3);
    // Four B fragment columns (N tiles n0 + 16j); lane-half picks K {0-7,16-23}/{8-15,24-31}.
    const __bf16* wp = W + (long)(n0 + sub) * ldw + (half << 3);

    v8f acc[2][4] = {};
    for (int kk = 0; kk < K; kk += 32) {
        __builtin_prefetch(ap0 + kk + 256, 0, 1);   // global_prefetch_b8
        __builtin_prefetch(ap1 + kk + 256, 0, 1);
        v16bf a0 = load_a_frag(ap0 + kk);
        v16bf a1 = load_a_frag(ap1 + kk);
#pragma unroll
        for (int j = 0; j < 4; ++j) {
            v16bf b = load_b_frag(wp + (long)(16 * j) * ldw + kk);
            acc[0][j] = __builtin_amdgcn_wmma_f32_16x16x32_bf16(
                false, a0, false, b, (short)0, acc[0][j], false, false);
            acc[1][j] = __builtin_amdgcn_wmma_f32_16x16x32_bf16(
                false, a1, false, b, (short)0, acc[1][j], false, false);
        }
    }

    // C/D layout: VGPR i -> row (m0 + 16*mi + 8*half + i), col (n0 + 16*j + sub)
#pragma unroll
    for (int mi = 0; mi < 2; ++mi) {
        const int rbase = m0 + 16 * mi + (half << 3);
#pragma unroll
        for (int j = 0; j < 4; ++j) {
            const int col  = n0 + 16 * j + sub;
            const float bv = bias ? bias[col] : 0.0f;
#pragma unroll
            for (int i = 0; i < 8; ++i) {
                long o  = (long)(rbase + i) * ldc + col;
                float v = acc[mi][j][i] + bv;
                if (Cf) Cf[o] = v;
                if (Cb) Cb[o] = f2bf(v);
            }
        }
    }
}

// ---------------------------------------------------------------------------
// Fused LSTM gate activation: gates = xih(+row-stride 0 for broadcast) + hWhh + b
// ---------------------------------------------------------------------------
__global__ __launch_bounds__(256)
void lstm_kernel(const float* __restrict__ xih, int xih_stride,
                 const float* __restrict__ hwhh, const float* __restrict__ bias,
                 float* __restrict__ h, float* __restrict__ c,
                 __bf16* __restrict__ hb)
{
    int idx = blockIdx.x * blockDim.x + threadIdx.x;
    if (idx >= Bn * Hn) return;
    int b = idx / Hn, j = idx - b * Hn;
    const float* xr = xih  + (long)b * xih_stride;
    const float* hr = hwhh + (long)b * G4H;
    float gi = xr[j]          + hr[j]          + bias[j];
    float gf = xr[Hn + j]     + hr[Hn + j]     + bias[Hn + j];
    float gg = xr[2 * Hn + j] + hr[2 * Hn + j] + bias[2 * Hn + j];
    float go = xr[3 * Hn + j] + hr[3 * Hn + j] + bias[3 * Hn + j];
    float cc = sigmoidf_(gf) * c[idx] + sigmoidf_(gi) * tanhf(gg);
    float hh = sigmoidf_(go) * tanhf(cc);
    c[idx] = cc;
    h[idx] = hh;
    hb[idx] = f2bf(hh);
}

// ---------------------------------------------------------------------------
// Glimpse attention (fused logits -> softmax -> weighted sum), one block per b.
// ref is bf16 (b,s,h); resident in L2 across all 128 decoder steps.
// ---------------------------------------------------------------------------
__global__ __launch_bounds__(256)
void glimpse_kernel(const float* __restrict__ qp, const __bf16* __restrict__ ref,
                    const float* __restrict__ V, float* __restrict__ q,
                    __bf16* __restrict__ qb)
{
    __shared__ float sl[Sn];
    __shared__ float sw[Sn];
    const int b = blockIdx.x, tid = threadIdx.x;
    const int lane = tid & 31, wv = tid >> 5;
    const __bf16* refb = ref + (size_t)b * Sn * Hn;
    const float*  qpb  = qp  + (size_t)b * Hn;

    for (int s = wv; s < Sn; s += 8) {
        const __bf16* rr = refb + (size_t)s * Hn;
        float acc = 0.0f;
        for (int hh = lane; hh < Hn; hh += 32)
            acc += V[hh] * tanhf(qpb[hh] + bf2f(rr[hh]));
        for (int o = 16; o > 0; o >>= 1) acc += __shfl_xor(acc, o, 32);
        if (lane == 0) sl[s] = acc;
    }
    __syncthreads();
    float m = -1e30f;
    for (int s = 0; s < Sn; ++s) m = fmaxf(m, sl[s]);
    float sum = 0.0f;
    for (int s = 0; s < Sn; ++s) sum += __expf(sl[s] - m);
    float inv = 1.0f / sum;
    if (tid < Sn) sw[tid] = __expf(sl[tid] - m) * inv;
    __syncthreads();
    for (int hh = tid; hh < Hn; hh += 256) {
        float acc = 0.0f;
        for (int s = 0; s < Sn; ++s)
            acc += sw[s] * bf2f(refb[(size_t)s * Hn + hh]);
        q[(size_t)b * Hn + hh]  = acc;
        qb[(size_t)b * Hn + hh] = f2bf(acc);
    }
}

// ---------------------------------------------------------------------------
// Pointer head: logits -> 10*tanh -> log_softmax -> -logp[target]/(B*S) into loss.
// ---------------------------------------------------------------------------
__global__ __launch_bounds__(256)
void pointer_loss_kernel(const float* __restrict__ qp, const __bf16* __restrict__ ref,
                         const float* __restrict__ V, const int* __restrict__ target,
                         float* __restrict__ loss)
{
    __shared__ float sl[Sn];
    const int b = blockIdx.x, tid = threadIdx.x;
    const int lane = tid & 31, wv = tid >> 5;
    const __bf16* refb = ref + (size_t)b * Sn * Hn;
    const float*  qpb  = qp  + (size_t)b * Hn;

    for (int s = wv; s < Sn; s += 8) {
        const __bf16* rr = refb + (size_t)s * Hn;
        float acc = 0.0f;
        for (int hh = lane; hh < Hn; hh += 32)
            acc += V[hh] * tanhf(qpb[hh] + bf2f(rr[hh]));
        for (int o = 16; o > 0; o >>= 1) acc += __shfl_xor(acc, o, 32);
        if (lane == 0) sl[s] = 10.0f * tanhf(acc);
    }
    __syncthreads();
    if (tid == 0) {
        float m = -1e30f;
        for (int s = 0; s < Sn; ++s) m = fmaxf(m, sl[s]);
        float sum = 0.0f;
        for (int s = 0; s < Sn; ++s) sum += __expf(sl[s] - m);
        int tgt = target[(size_t)b * Sn];           // target[:, 0]
        float logp = sl[tgt] - m - __logf(sum);
        atomicAdd(loss, -logp * (1.0f / ((float)Bn * (float)Sn)));
    }
}

// ---------------------------------------------------------------------------
// Small utilities
// ---------------------------------------------------------------------------
__global__ void cvt_f32_bf16(const float* __restrict__ s, __bf16* __restrict__ d, int n) {
    int i = blockIdx.x * blockDim.x + threadIdx.x;
    if (i < n) d[i] = f2bf(s[i]);
}
__global__ void zero_f32(float* p, int n) {
    int i = blockIdx.x * blockDim.x + threadIdx.x;
    if (i < n) p[i] = 0.0f;
}
__global__ void zero_bf16(__bf16* p, int n) {
    int i = blockIdx.x * blockDim.x + threadIdx.x;
    if (i < n) p[i] = f2bf(0.0f);
}
// xih0[j] = dec_start . dec_Wih[j,:]   (tiny, fp32 exact)
__global__ void vecproj_kernel(const float* __restrict__ x, const float* __restrict__ W,
                               float* __restrict__ out) {
    int j = blockIdx.x * blockDim.x + threadIdx.x;
    if (j >= G4H) return;
    const float* wr = W + (long)j * En;
    float s = 0.0f;
    for (int k = 0; k < En; ++k) s += x[k] * wr[k];
    out[j] = s;
}

// ---------------------------------------------------------------------------
extern "C" void kernel_launch(void* const* d_in, const int* in_sizes, int n_in,
                              void* d_out, int out_size, void* d_ws, size_t ws_size,
                              hipStream_t stream)
{
    const int*   inputs    = (const int*)  d_in[0];
    const int*   target    = (const int*)  d_in[1];
    const float* embedding = (const float*)d_in[2];
    const float* enc_Wih   = (const float*)d_in[3];
    const float* enc_Whh   = (const float*)d_in[4];
    const float* enc_b     = (const float*)d_in[5];
    const float* dec_Wih   = (const float*)d_in[6];
    const float* dec_Whh   = (const float*)d_in[7];
    const float* dec_b     = (const float*)d_in[8];
    const float* g_Wq      = (const float*)d_in[9];
    const float* g_bq      = (const float*)d_in[10];
    const float* g_Wref    = (const float*)d_in[11];
    const float* g_bref    = (const float*)d_in[12];
    const float* g_V       = (const float*)d_in[13];
    const float* p_Wq      = (const float*)d_in[14];
    const float* p_bq      = (const float*)d_in[15];
    const float* p_Wref    = (const float*)d_in[16];
    const float* p_bref    = (const float*)d_in[17];
    const float* p_V       = (const float*)d_in[18];
    const float* dec_start = (const float*)d_in[19];
    float* loss = (float*)d_out;

    char* base = (char*)d_ws;
    size_t off = 0;
    auto alloc = [&](size_t bytes) -> char* {
        off = (off + 255) & ~(size_t)255;
        char* p = base + off;
        off += bytes;
        return p;
    };

    __bf16* emb_bf    = (__bf16*)alloc((size_t)Sn * En * 2);
    __bf16* eWih_bf   = (__bf16*)alloc((size_t)G4H * En * 2);
    __bf16* eWhh_bf   = (__bf16*)alloc((size_t)G4H * Hn * 2);
    __bf16* dWih_bf   = (__bf16*)alloc((size_t)G4H * En * 2);
    __bf16* dWhh_bf   = (__bf16*)alloc((size_t)G4H * Hn * 2);
    __bf16* gWq_bf    = (__bf16*)alloc((size_t)Hn * Hn * 2);
    __bf16* gWref_bf  = (__bf16*)alloc((size_t)Hn * Hn * 2);
    __bf16* pWq_bf    = (__bf16*)alloc((size_t)Hn * Hn * 2);
    __bf16* pWref_bf  = (__bf16*)alloc((size_t)Hn * Hn * 2);
    __bf16* refg_bf   = (__bf16*)alloc((size_t)Bn * Sn * Hn * 2);   // 32 MB (L2-resident)
    __bf16* refp_bf   = (__bf16*)alloc((size_t)Bn * Sn * Hn * 2);   // 32 MB (L2-resident)
    float*  hbuf      = (float*) alloc((size_t)Bn * Hn * 4);
    float*  cbuf      = (float*) alloc((size_t)Bn * Hn * 4);
    __bf16* hb        = (__bf16*)alloc((size_t)Bn * Hn * 2);
    float*  xih       = (float*) alloc((size_t)Bn * G4H * 4);
    float*  hwhh      = (float*) alloc((size_t)Bn * G4H * 4);
    float*  xihj      = (float*) alloc((size_t)Bn * G4H * 4);
    float*  xih0      = (float*) alloc((size_t)G4H * 4);
    float*  qp        = (float*) alloc((size_t)Bn * Hn * 4);
    float*  qbuf      = (float*) alloc((size_t)Bn * Hn * 4);
    __bf16* qb        = (__bf16*)alloc((size_t)Bn * Hn * 2);
    (void)ws_size; (void)n_in; (void)in_sizes;

    auto cvt = [&](const float* s, __bf16* d, int n) {
        cvt_f32_bf16<<<(n + 255) / 256, 256, 0, stream>>>(s, d, n);
    };
    auto gemm = [&](const __bf16* A, long lda, const int* gth, int gs,
                    const __bf16* W, long ldw, const float* bias,
                    float* Cf, __bf16* Cb, long ldc, int M, int N, int K) {
        dim3 grid(M / 32, N / 256);     // 32x64 macro-tile per wave, 4 waves/block
        gemm_bf16_kernel<<<grid, 128, 0, stream>>>(A, lda, gth, gs, W, ldw, bias,
                                                   Cf, Cb, ldc, K);
    };

    // --- one-time conversions + state init ---------------------------------
    cvt(embedding, emb_bf, Sn * En);
    cvt(enc_Wih, eWih_bf, G4H * En);
    cvt(enc_Whh, eWhh_bf, G4H * Hn);
    cvt(dec_Wih, dWih_bf, G4H * En);
    cvt(dec_Whh, dWhh_bf, G4H * Hn);
    cvt(g_Wq,   gWq_bf,   Hn * Hn);
    cvt(g_Wref, gWref_bf, Hn * Hn);
    cvt(p_Wq,   pWq_bf,   Hn * Hn);
    cvt(p_Wref, pWref_bf, Hn * Hn);
    zero_f32 <<<(Bn * Hn + 255) / 256, 256, 0, stream>>>(hbuf, Bn * Hn);
    zero_f32 <<<(Bn * Hn + 255) / 256, 256, 0, stream>>>(cbuf, Bn * Hn);
    zero_bf16<<<(Bn * Hn + 255) / 256, 256, 0, stream>>>(hb, Bn * Hn);
    zero_f32 <<<1, 32, 0, stream>>>(loss, out_size);

    // --- encoder: 128 recurrent steps, ref rows emitted per step -----------
    for (int t = 0; t < Sn; ++t) {
        // xih = embedding[inputs[:,t]] @ enc_Wih^T   (gather-GEMM)
        gemm(emb_bf, En, inputs + t, Sn, eWih_bf, En, nullptr,
             xih, nullptr, G4H, Bn, G4H, En);
        // hwhh = h @ enc_Whh^T
        gemm(hb, Hn, nullptr, 0, eWhh_bf, Hn, nullptr,
             hwhh, nullptr, G4H, Bn, G4H, Hn);
        lstm_kernel<<<(Bn * Hn + 255) / 256, 256, 0, stream>>>(
            xih, G4H, hwhh, enc_b, hbuf, cbuf, hb);
        // ref_g[:, :, t] and ref_p[:, :, t]  (stored (b,s,h) as bf16)
        gemm(hb, Hn, nullptr, 0, gWref_bf, Hn, g_bref,
             nullptr, refg_bf + (size_t)t * Hn, (long)Sn * Hn, Bn, Hn, Hn);
        gemm(hb, Hn, nullptr, 0, pWref_bf, Hn, p_bref,
             nullptr, refp_bf + (size_t)t * Hn, (long)Sn * Hn, Bn, Hn, Hn);
    }

    // --- decoder precompute -------------------------------------------------
    vecproj_kernel<<<(G4H + 255) / 256, 256, 0, stream>>>(dec_start, dec_Wih, xih0);
    // xihj = embedding[target[:,0]] @ dec_Wih^T
    gemm(emb_bf, En, target, Sn, dWih_bf, En, nullptr,
         xihj, nullptr, G4H, Bn, G4H, En);

    // --- decoder: 128 steps with glimpse + pointer attention ----------------
    for (int t = 0; t < Sn; ++t) {
        gemm(hb, Hn, nullptr, 0, dWhh_bf, Hn, nullptr,
             hwhh, nullptr, G4H, Bn, G4H, Hn);
        if (t == 0)
            lstm_kernel<<<(Bn * Hn + 255) / 256, 256, 0, stream>>>(
                xih0, 0, hwhh, dec_b, hbuf, cbuf, hb);      // broadcast row
        else
            lstm_kernel<<<(Bn * Hn + 255) / 256, 256, 0, stream>>>(
                xihj, G4H, hwhh, dec_b, hbuf, cbuf, hb);
        // glimpse: qp = h @ g_Wq^T + g_bq ; softmax attention over ref_g
        gemm(hb, Hn, nullptr, 0, gWq_bf, Hn, g_bq,
             qp, nullptr, Hn, Bn, Hn, Hn);
        glimpse_kernel<<<Bn, 256, 0, stream>>>(qp, refg_bf, g_V, qbuf, qb);
        // pointer: qp = q @ p_Wq^T + p_bq ; loss over ref_p
        gemm(qb, Hn, nullptr, 0, pWq_bf, Hn, p_bq,
             qp, nullptr, Hn, Bn, Hn, Hn);
        pointer_loss_kernel<<<Bn, 256, 0, stream>>>(qp, refp_bf, p_V, target, loss);
    }
}
